// DPN_33646773797026
// MI455X (gfx1250) — matrix operand order repaired and checked by reading, not compiled, not run on hardware
//
#include <hip/hip_runtime.h>

#define HH   128
#define WWD  128
#define CIN  64
#define OUTC 64
#define BB   2
#define NPT  9
#define KK   576           // CIN * 9
#define KSTEPS 18          // KK / 32
#define HWSZ (HH*WWD)      // 16384
#define POSTOT (BB*HWSZ)   // 32768
#define POS_PER_BLOCK 32   // == wave size: lane <-> position in gather stage
#define ROWPAD 584         // 576 + 8 f16 pad to de-periodize LDS banks

typedef __attribute__((ext_vector_type(16))) _Float16 v16h;
typedef __attribute__((ext_vector_type(8)))  _Float16 v8h;
typedef __attribute__((ext_vector_type(8)))  float    v8f;

// ---------------------------------------------------------------------------
// Kernel 0: convert Wc (f32, layout [o][c][3][3] == [o][k]) to f16 once.
// ---------------------------------------------------------------------------
__global__ void wc_to_f16_kernel(const float* __restrict__ Wc,
                                 _Float16* __restrict__ wch) {
    int i = blockIdx.x * 256 + threadIdx.x;
    if (i < OUTC * KK) wch[i] = (_Float16)Wc[i];
}

// ---------------------------------------------------------------------------
// Kernel 1: offset conv (3x3, C=64 -> 18). Wp cached in LDS (41 KB),
// wave-uniform LDS reads broadcast. One thread per pixel.
// ---------------------------------------------------------------------------
__global__ void __launch_bounds__(256) offset_conv_kernel(
        const float* __restrict__ x, const float* __restrict__ Wp,
        const float* __restrict__ bp, float* __restrict__ off) {
    __shared__ float sWp[18 * KK];   // [j][c*9 + t]
    __shared__ float sBp[18];
    int tid = threadIdx.x;
    for (int i = tid; i < 18 * KK; i += 256) sWp[i] = Wp[i];
    if (tid < 18) sBp[tid] = bp[tid];
    __syncthreads();

    int p  = blockIdx.x * 256 + tid;          // linear (b,h,w)
    int b  = p >> 14;
    int hw = p & (HWSZ - 1);
    int h  = hw >> 7, w = hw & 127;

    float acc[18];
    #pragma unroll
    for (int j = 0; j < 18; ++j) acc[j] = sBp[j];

    const float* xb = x + b * CIN * HWSZ;
    for (int c = 0; c < CIN; ++c) {
        float xv[9];
        #pragma unroll
        for (int u = 0; u < 3; ++u)
            #pragma unroll
            for (int v = 0; v < 3; ++v) {
                int hh = h + u - 1, ww = w + v - 1;
                bool in = (hh >= 0) & (hh < HH) & (ww >= 0) & (ww < WWD);
                xv[u*3 + v] = in ? xb[c * HWSZ + hh * WWD + ww] : 0.0f;
            }
        const float* wrow = sWp + c * 9;
        #pragma unroll
        for (int j = 0; j < 18; ++j) {
            float a = acc[j];
            #pragma unroll
            for (int t = 0; t < 9; ++t) a = fmaf(wrow[j * KK + t], xv[t], a);
            acc[j] = a;
        }
    }
    float* ob = off + b * 18 * HWSZ + hw;
    #pragma unroll
    for (int j = 0; j < 18; ++j) ob[j * HWSZ] = acc[j];
}

// ---------------------------------------------------------------------------
// Kernel 2: bilinear gather + WMMA GEMM out[64 x pos] = Wc[64x576]*samp[576xpos]
// Block: 256 threads / 8 waves, 32 positions (== 1 wave of lanes for gathers).
// Gather stage: lane = position -> near-coalesced corner loads (offsets are
// O(0.1), so consecutive w positions hit consecutive addresses). OOB corners
// use address 0 with weight 0 => unconditional, clause-able loads.
// ---------------------------------------------------------------------------
__global__ void __launch_bounds__(256) dcn_main_kernel(
        const float* __restrict__ x, const float* __restrict__ off,
        const _Float16* __restrict__ wch, float* __restrict__ out) {
    __shared__ _Float16 sB[POS_PER_BLOCK * ROWPAD];          // 37376 B
    __shared__ int      sAddr[NPT * 4 * POS_PER_BLOCK];      //  4608 B [n][k][pp]
    __shared__ float    sWt[NPT * 4 * POS_PER_BLOCK];        //  4608 B [n][k][pp]

    int tid   = threadIdx.x;
    int pbase = blockIdx.x * POS_PER_BLOCK;

    // ---- Stage 1: sampling geometry, one entry per (pos, n) ----
    for (int combo = tid; combo < POS_PER_BLOCK * NPT; combo += 256) {
        int pp = combo / NPT, n = combo - pp * NPT;
        int p  = pbase + pp;
        int b  = p >> 14;
        int hw = p & (HWSZ - 1);
        int h  = hw >> 7, w = hw & 127;

        float ox = off[b * 18 * HWSZ + n * HWSZ + hw];
        float oy = off[b * 18 * HWSZ + (n + NPT) * HWSZ + hw];
        int   kx = n / 3, ky = n - kx * 3;
        float px = ox + (float)(kx - 1) + (float)(h + 1);   // padded coords
        float py = oy + (float)(ky - 1) + (float)(w + 1);

        float fx = floorf(px), fy = floorf(py);
        int qltx = min(max((int)fx, 0), HH + 1);
        int qlty = min(max((int)fy, 0), WWD + 1);
        int qrbx = min(max((int)fx + 1, 0), HH + 1);
        int qrby = min(max((int)fy + 1, 0), WWD + 1);
        float pxc = fminf(fmaxf(px, 0.0f), (float)(HH + 1));
        float pyc = fminf(fmaxf(py, 0.0f), (float)(WWD + 1));

        float wlx = 1.0f + ((float)qltx - pxc);
        float wly = 1.0f + ((float)qlty - pyc);
        float wrx = 1.0f - ((float)qrbx - pxc);
        float wry = 1.0f - ((float)qrby - pyc);

        int ixs[4] = {qltx, qrbx, qltx, qrbx};   // lt, rb, lb, rt
        int iys[4] = {qlty, qrby, qrby, qlty};
        float wt[4] = {wlx * wly, wrx * wry, wlx * wry, wrx * wly};

        int base = b * CIN * HWSZ;
        #pragma unroll
        for (int k = 0; k < 4; ++k) {
            bool in = (ixs[k] >= 1) & (ixs[k] <= HH) &
                      (iys[k] >= 1) & (iys[k] <= WWD);
            int slot = (n * 4 + k) * POS_PER_BLOCK + pp;   // lane-unit-stride
            sAddr[slot] = in ? (base + (ixs[k] - 1) * WWD + (iys[k] - 1)) : 0;
            sWt[slot]   = in ? wt[k] : 0.0f;
        }
    }
    __syncthreads();

    // ---- Stage 2: lane = position; wave iterates (c,n) pairs ----
    {
        int wave = tid >> 5, lane = tid & 31;
        for (int pair = wave; pair < CIN * NPT; pair += 8) {
            int c = pair / NPT, n = pair - c * NPT;       // pair = c*9 + n
            int coff = c * HWSZ;
            float v = 0.0f;
            #pragma unroll
            for (int k = 0; k < 4; ++k) {
                int slot  = (n * 4 + k) * POS_PER_BLOCK + lane;
                int   a   = sAddr[slot];
                float g   = sWt[slot];
                v = fmaf(g, x[a + coff], v);              // coalesced across lanes
            }
            sB[lane * ROWPAD + c * NPT + n] = (_Float16)v;  // k-index = c*9+n
        }
    }
    __syncthreads();

    // ---- Stage 3: per-wave 16x16 WMMA tile, K = 576 in 18 steps ----
    int wave = tid >> 5, lane = tid & 31;
    int mtile = wave & 3, ptile = wave >> 2;
    int lo16 = lane & 15, hi = lane >> 4;

    // A (16-bit, 16x32): lane hi-half holds K = hi*8+{0..7} and 16+hi*8+{0..7}
    const _Float16* aBase = wch + (mtile * 16 + lo16) * KK + hi * 8;
    // B (16-bit, 32x16): lane hi-half holds K = hi*8+{0..7} and 16+hi*8+{0..7}
    const _Float16* bBase = sB + (ptile * 16 + lo16) * ROWPAD + hi * 8;

    v8f acc = {};
    #pragma unroll 2
    for (int ks = 0; ks < KSTEPS; ++ks) {
        v8h a0 = *(const v8h*)(aBase + ks * 32);
        v8h a1 = *(const v8h*)(aBase + ks * 32 + 16);
        v8h b0 = *(const v8h*)(bBase + ks * 32);
        v8h b1 = *(const v8h*)(bBase + ks * 32 + 16);
        v16h av = __builtin_shufflevector(a0, a1,
                    0,1,2,3,4,5,6,7,8,9,10,11,12,13,14,15);
        v16h bv = __builtin_shufflevector(b0, b1,
                    0,1,2,3,4,5,6,7,8,9,10,11,12,13,14,15);
        acc = __builtin_amdgcn_wmma_f32_16x16x32_f16(
                  false, av, false, bv, (short)0, acc, false, false);
    }

    // ---- Store: D layout — lane = column(pos), VGPR r = row hi*8+r ----
    int pcol = pbase + ptile * 16 + lo16;
    int b    = pcol >> 14;
    int hw   = pcol & (HWSZ - 1);
    float* ob = out + b * OUTC * HWSZ + hw;
    #pragma unroll
    for (int r = 0; r < 8; ++r) {
        int o = mtile * 16 + hi * 8 + r;
        ob[o * HWSZ] = acc[r];
    }
}

// ---------------------------------------------------------------------------
// Launch: inputs are x, Wp, bp, Wc (all f32). Workspace layout:
//   [0, 2359296)          : f32 offset map  [2][18][128][128]
//   [2359296, +73728)     : f16 Wc          [64][576]
// ---------------------------------------------------------------------------
extern "C" void kernel_launch(void* const* d_in, const int* in_sizes, int n_in,
                              void* d_out, int out_size, void* d_ws, size_t ws_size,
                              hipStream_t stream) {
    (void)in_sizes; (void)n_in; (void)out_size; (void)ws_size;
    const float* x  = (const float*)d_in[0];
    const float* Wp = (const float*)d_in[1];
    const float* bp = (const float*)d_in[2];
    const float* Wc = (const float*)d_in[3];
    float* out = (float*)d_out;

    float*    d_off = (float*)d_ws;
    _Float16* d_wch = (_Float16*)((char*)d_ws + (size_t)BB * 18 * HWSZ * sizeof(float));

    wc_to_f16_kernel<<<(OUTC * KK + 255) / 256, 256, 0, stream>>>(Wc, d_wch);
    offset_conv_kernel<<<POSTOT / 256, 256, 0, stream>>>(x, Wp, bp, d_off);
    dcn_main_kernel<<<POSTOT / POS_PER_BLOCK, 256, 0, stream>>>(x, d_off, d_wch, out);
}
